// GRUReg_55336358641941
// MI455X (gfx1250) — matrix operand order repaired and checked by reading, not compiled, not run on hardware
//
#include <hip/hip_runtime.h>
#include <math.h>

#define B_   16
#define N_   2048
#define HID_ 64
#define EPS_ 1e-5f

typedef unsigned short u16;
typedef unsigned int   u32;
typedef __attribute__((ext_vector_type(4)))  u32    u32x4;
typedef __attribute__((ext_vector_type(8)))  float  v8f;
typedef __attribute__((ext_vector_type(16))) __bf16 v16bf;

// ---------- bf16 helpers (round-to-nearest-even) ----------
static __device__ __forceinline__ u16 f2bf(float f) {
  union { float f; u32 u; } v; v.f = f;
  u32 u = v.u + 0x7FFFu + ((v.u >> 16) & 1u);
  return (u16)(u >> 16);
}
static __device__ __forceinline__ float bf2f(u16 h) {
  union { u32 u; float f; } v; v.u = ((u32)h) << 16; return v.f;
}
static __device__ __forceinline__ float sigmoidf_(float x) { return 1.0f / (1.0f + expf(-x)); }

// ============================================================
// KNN: per-batch xyz (24KB) cached in LDS, sorted top-16 in registers.
// ============================================================
__global__ void knn16_kernel(const float* __restrict__ pc, int* __restrict__ idx) {
  __shared__ float sx[N_], sy[N_], sz[N_];
  int b = blockIdx.y;
  const float* px = pc + (size_t)b * 3 * N_;
  for (int i = threadIdx.x; i < N_; i += blockDim.x) {
    sx[i] = px[i]; sy[i] = px[N_ + i]; sz[i] = px[2 * N_ + i];
  }
  __syncthreads();
  int n = blockIdx.x * blockDim.x + threadIdx.x;
  if (n >= N_) return;
  float qx = sx[n], qy = sy[n], qz = sz[n];
  float bd[16]; int bi[16];
#pragma unroll
  for (int t = 0; t < 16; t++) { bd[t] = 3.4e38f; bi[t] = n; }
  for (int m = 0; m < N_; m++) {
    float dx = qx - sx[m], dy = qy - sy[m], dz = qz - sz[m];
    float d = dx * dx + dy * dy + dz * dz;
    if (d < bd[15]) {
      float cd = d; int ci = m;
#pragma unroll
      for (int t = 0; t < 16; t++) {
        if (cd < bd[t]) {
          float td = bd[t]; int ti = bi[t];
          bd[t] = cd; bi[t] = ci; cd = td; ci = ti;
        }
      }
    }
  }
  int* out = idx + ((size_t)b * N_ + n) * 16;
#pragma unroll
  for (int t = 0; t < 16; t++) out[t] = bi[t];
}

// ============================================================
// Weight pack: f32 [cin x cout] -> bf16-pair dwords in B-fragment order.
// Layout: dword index = ((s*TN + nt)*32 + lane)*8 + j
//   where s = K/32 step, nt = N-tile, lane = half*16+lm,
//   value = bf16(W[32s+8*half+2*(j&3)+16*(j>>2)][nt*16+lm]) | bf16(W[..+1][..])<<16
// This makes the GEMM's B fragment two ds_load_b128's, zero VALU packing.
// ============================================================
__global__ void wpack_kernel(const float* __restrict__ w, u32* __restrict__ wp,
                             int cin, int cout, int Kpad) {
  int i = blockIdx.x * blockDim.x + threadIdx.x;
  int TN = cout >> 4;
  int total = (Kpad >> 5) * TN * 256;
  if (i >= total) return;
  int j = i & 7, lane = (i >> 3) & 31, rest = i >> 8;
  int nt = rest % TN, s = rest / TN;
  int half = lane >> 4, lm = lane & 15;
  int kb = 32 * s + 8 * half + 2 * (j & 3) + 16 * (j >> 2);
  int n = nt * 16 + lm;
  u32 lo = (kb     < cin) ? (u32)f2bf(w[kb * cout + n])       : 0u;
  u32 hi = (kb + 1 < cin) ? (u32)f2bf(w[(kb + 1) * cout + n]) : 0u;
  wp[i] = lo | (hi << 16);
}

// ============================================================
// WMMA GEMM: X[M x Cout] = A(bf16)[M x Kpad] * W + bias.
// One wave per 16 rows, TN=Cout/16 accumulator tiles per wave.
// A fragment  = 2 x global_load_b128 (16-bit A 16x32 layout is dword-contiguous)
// B fragments = 2 x ds_load_b128 each from pre-packed LDS copy
// ============================================================
template <int TN>
__global__ void wmma_gemm_kernel(const u32* __restrict__ Au, const u32* __restrict__ Wp,
                                 const float* __restrict__ bias, float* __restrict__ X,
                                 int M, int Kpad) {
  constexpr int Cout = TN * 16;
  extern __shared__ u32 sWp[];
  int tw = (Kpad >> 5) * TN * 256;
  for (int i = threadIdx.x; i < tw; i += blockDim.x) sWp[i] = Wp[i];
  __syncthreads();

  int mt = blockIdx.x * (blockDim.x >> 5) + (threadIdx.x >> 5);
  if (mt * 16 >= M) return;
  int lane = threadIdx.x & 31;
  int half = lane >> 4, lm = lane & 15;
  int row = mt * 16 + lm;
  size_t arow = (size_t)row * (Kpad >> 1);

  v8f acc[TN];
#pragma unroll
  for (int t = 0; t < TN; t++) acc[t] = (v8f){0.f, 0.f, 0.f, 0.f, 0.f, 0.f, 0.f, 0.f};

  for (int k0 = 0, s = 0; k0 < Kpad; k0 += 32, s++) {
    union { u32x4 q[2]; v16bf v; } a;
    const u32* ap = Au + arow + (k0 >> 1) + 4 * half;
    a.q[0] = *(const u32x4*)(ap);
    a.q[1] = *(const u32x4*)(ap + 8);
#pragma unroll
    for (int t = 0; t < TN; t++) {
      union { u32x4 q[2]; v16bf v; } bfv;
      const u32* bp = sWp + (((s * TN + t) * 32 + lane) << 3);
      bfv.q[0] = *(const u32x4*)(bp);
      bfv.q[1] = *(const u32x4*)(bp + 4);
      acc[t] = __builtin_amdgcn_wmma_f32_16x16x32_bf16(false, a.v, false, bfv.v,
                                                       (short)0, acc[t], false, false);
    }
  }

#pragma unroll
  for (int t = 0; t < TN; t++) {
    int ncol = t * 16 + lm;
    float bv = bias[ncol];
#pragma unroll
    for (int i = 0; i < 8; i++) {
      int r = mt * 16 + i + 8 * half;   // C/D layout: M = i + 8*half, N = lane%16
      X[(size_t)r * Cout + ncol] = acc[t][i] + bv;
    }
  }
}

// ============================================================
// Batch-global per-channel stats over N*k rows: mean + invstd
// ============================================================
__global__ void stats_kernel(const float* __restrict__ X, float* __restrict__ stats,
                             int rows, int Cout) {
  __shared__ float ssum[256], ssq[256];
  int b = blockIdx.x / Cout, c = blockIdx.x - b * Cout;
  const float* p = X + (size_t)b * rows * Cout + c;
  float s = 0.f, q = 0.f;
  for (int r = threadIdx.x; r < rows; r += blockDim.x) {
    float v = p[(size_t)r * Cout]; s += v; q += v * v;
  }
  ssum[threadIdx.x] = s; ssq[threadIdx.x] = q;
  __syncthreads();
  for (int o = 128; o > 0; o >>= 1) {
    if ((int)threadIdx.x < o) {
      ssum[threadIdx.x] += ssum[threadIdx.x + o];
      ssq[threadIdx.x]  += ssq[threadIdx.x + o];
    }
    __syncthreads();
  }
  if (threadIdx.x == 0) {
    float m   = ssum[0] / (float)rows;
    float var = ssq[0] / (float)rows - m * m;
    stats[2 * blockIdx.x]     = m;
    stats[2 * blockIdx.x + 1] = rsqrtf(var + EPS_);
  }
}

// ============================================================
// Normalize (+optional ReLU), emit bf16 A-matrix for next layer (zero pad)
// ============================================================
__global__ void norm_kernel(const float* __restrict__ X, const float* __restrict__ stats,
                            u16* __restrict__ Aout, int rowsPerBatch, int Cout, int KpadOut,
                            long long total, int relu) {
  long long i = (long long)blockIdx.x * blockDim.x + threadIdx.x;
  if (i >= total) return;
  long long row = i / KpadOut;
  int c = (int)(i - row * KpadOut);
  float v = 0.f;
  if (c < Cout) {
    int b = (int)(row / rowsPerBatch);
    float m   = stats[2 * (b * Cout + c)];
    float inv = stats[2 * (b * Cout + c) + 1];
    v = (X[row * (long long)Cout + c] - m) * inv;
    if (relu) v = fmaxf(v, 0.f);
  }
  Aout[i] = f2bf(v);
}

// ============================================================
// Max over k neighbors: bf16 [M x Kpad] -> f32 [B*N x C]
// ============================================================
__global__ void maxk_kernel(const u16* __restrict__ An, float* __restrict__ out,
                            int k, int C, int Kpad, long long total) {
  long long i = (long long)blockIdx.x * blockDim.x + threadIdx.x;
  if (i >= total) return;
  long long bn = i / C; int c = (int)(i - bn * C);
  float m = -3.4e38f;
  for (int kk = 0; kk < k; kk++)
    m = fmaxf(m, bf2f(An[(bn * k + kk) * (long long)Kpad + c]));
  out[i] = m;
}

// ============================================================
// Gather kernels (build bf16 A rows for layer-0 of each SA stage)
// ============================================================
__global__ void gather_fp1_kernel(const float* __restrict__ pc, const float* __restrict__ flow,
                                  const int* __restrict__ idx, u16* __restrict__ A) {
  long long r = (long long)blockIdx.x * blockDim.x + threadIdx.x;
  long long total = (long long)B_ * N_ * 16;
  if (r >= total) return;
  int kk = (int)(r & 15);
  long long bn = r >> 4;
  int n = (int)(bn % N_);
  int b = (int)(bn / N_);
  int j = idx[bn * 16 + kk];
  const float* px = pc   + (size_t)b * 3 * N_;
  const float* fl = flow + (size_t)b * 3 * N_;
  u16* a = A + r * 32;
  a[0] = f2bf(px[j]          - px[n]);
  a[1] = f2bf(px[N_ + j]     - px[N_ + n]);
  a[2] = f2bf(px[2 * N_ + j] - px[2 * N_ + n]);
  a[3] = f2bf(fl[j]); a[4] = f2bf(fl[N_ + j]); a[5] = f2bf(fl[2 * N_ + j]);
#pragma unroll
  for (int c = 6; c < 32; c++) a[c] = 0;
}

__global__ void gather_fp2_kernel(const float* __restrict__ pc, const float* __restrict__ ff1,
                                  const int* __restrict__ idx, u16* __restrict__ A) {
  long long r = (long long)blockIdx.x * blockDim.x + threadIdx.x;
  long long total = (long long)B_ * N_ * 8;
  if (r >= total) return;
  int kk = (int)(r & 7);
  long long bn = r >> 3;
  int n = (int)(bn % N_);
  int b = (int)(bn / N_);
  int j = idx[bn * 16 + kk];   // first 8 of the sorted 16
  const float* px = pc + (size_t)b * 3 * N_;
  u16* a = A + r * 64;
  a[0] = f2bf(px[j]          - px[n]);
  a[1] = f2bf(px[N_ + j]     - px[N_ + n]);
  a[2] = f2bf(px[2 * N_ + j] - px[2 * N_ + n]);
  long long bj = (long long)b * N_ + j;
  for (int c = 0; c < 32; c++) a[3 + c] = f2bf(ff1[bj * 32 + c]);
  for (int c = 35; c < 64; c++) a[c] = 0;
}

__global__ void gather_gate_kernel(const float* __restrict__ pc, const float* __restrict__ hsrc,
                                   int hIsCN,
                                   const float* __restrict__ f1n, const float* __restrict__ cost,
                                   const float* __restrict__ ff2, const float* __restrict__ flow,
                                   const int* __restrict__ idx, u16* __restrict__ A) {
  long long r = (long long)blockIdx.x * blockDim.x + threadIdx.x;
  long long total = (long long)B_ * N_ * 4;
  if (r >= total) return;
  int kk = (int)(r & 3);
  long long bn = r >> 2;
  int n = (int)(bn % N_);
  int b = (int)(bn / N_);
  int j = idx[bn * 16 + kk];   // first 4 of the sorted 16
  const float* px = pc   + (size_t)b * 3 * N_;
  const float* fl = flow + (size_t)b * 3 * N_;
  long long bj = (long long)b * N_ + j;
  u16* a = A + r * 224;
  a[0] = f2bf(px[j]          - px[n]);
  a[1] = f2bf(px[N_ + j]     - px[N_ + n]);
  a[2] = f2bf(px[2 * N_ + j] - px[2 * N_ + n]);
  for (int c = 0; c < 64; c++)
    a[3 + c] = f2bf(hIsCN ? hsrc[((size_t)b * 64 + c) * N_ + j] : hsrc[bj * 64 + c]);
  for (int c = 0; c < 64; c++) a[67 + c]  = f2bf(f1n[((size_t)b * 64 + c) * N_ + j]);
  for (int c = 0; c < 64; c++) a[131 + c] = f2bf(cost[((size_t)b * 64 + c) * N_ + j]);
  for (int c = 0; c < 16; c++) a[195 + c] = f2bf(ff2[bj * 16 + c]);
  a[211] = f2bf(fl[j]); a[212] = f2bf(fl[N_ + j]); a[213] = f2bf(fl[2 * N_ + j]);
#pragma unroll
  for (int c = 214; c < 224; c++) a[c] = 0;
}

// r*h (rh stored [B][N][64]; h is [B][64][N])
__global__ void rh_kernel(const float* __restrict__ rpre, const float* __restrict__ h,
                          float* __restrict__ rh, long long total) {
  long long i = (long long)blockIdx.x * blockDim.x + threadIdx.x;
  if (i >= total) return;
  long long bn = i >> 6; int c = (int)(i & 63);
  int n = (int)(bn % N_);
  int b = (int)(bn / N_);
  rh[i] = sigmoidf_(rpre[i]) * h[((size_t)b * 64 + c) * N_ + n];
}

// out[b,c,n] = (1-z)*h + z*tanh(q)
__global__ void combine_kernel(const float* __restrict__ zpre, const float* __restrict__ qpre,
                               const float* __restrict__ h, float* __restrict__ out,
                               long long total) {
  long long i = (long long)blockIdx.x * blockDim.x + threadIdx.x;
  if (i >= total) return;
  long long bc = i / N_; int n = (int)(i - bc * N_);
  int c = (int)(bc & 63);
  int b = (int)(bc >> 6);
  long long ip = (((long long)b * N_ + n) << 6) + c;
  float z = sigmoidf_(zpre[ip]);
  float q = tanhf(qpre[ip]);
  out[i] = (1.f - z) * h[i] + z * q;
}

// ============================================================
// Host-side helpers
// ============================================================
static void launch_gemm(const u16* A, const u32* Wp, const float* bias, float* X,
                        int M, int Kpad, int Cout, hipStream_t stream) {
  unsigned blocks = (unsigned)((M / 16 + 7) / 8);
  size_t smem = (size_t)(Kpad >> 5) * (Cout >> 4) * 256 * sizeof(u32);
  const u32* Au = (const u32*)A;
  if (Cout == 16)
    wmma_gemm_kernel<1><<<blocks, 256, smem, stream>>>(Au, Wp, bias, X, M, Kpad);
  else if (Cout == 32)
    wmma_gemm_kernel<2><<<blocks, 256, smem, stream>>>(Au, Wp, bias, X, M, Kpad);
  else
    wmma_gemm_kernel<4><<<blocks, 256, smem, stream>>>(Au, Wp, bias, X, M, Kpad);
}

static void layer_forward(const u16* A, const float* w, const float* bias, u32* Wp,
                          float* X, float* stats, u16* Aout,
                          int M, int cin, int Kpad, int Cout, int KpadOut,
                          int rowsPerBatch, int relu, hipStream_t stream) {
  int tp = (Kpad >> 5) * (Cout >> 4) * 256;
  wpack_kernel<<<(tp + 255) / 256, 256, 0, stream>>>(w, Wp, cin, Cout, Kpad);
  launch_gemm(A, Wp, bias, X, M, Kpad, Cout, stream);
  stats_kernel<<<B_ * Cout, 256, 0, stream>>>(X, stats, rowsPerBatch, Cout);
  long long total = (long long)M * KpadOut;
  norm_kernel<<<(unsigned)((total + 255) / 256), 256, 0, stream>>>(
      X, stats, Aout, rowsPerBatch, Cout, KpadOut, total, relu);
}

extern "C" void kernel_launch(void* const* d_in, const int* in_sizes, int n_in,
                              void* d_out, int out_size, void* d_ws, size_t ws_size,
                              hipStream_t stream) {
  const float* pc   = (const float*)d_in[0];
  const float* h    = (const float*)d_in[1];
  const float* f1n  = (const float*)d_in[2];
  const float* cost = (const float*)d_in[3];
  const float* flow = (const float*)d_in[4];
  const float* fp1_w[3] = { (const float*)d_in[5], (const float*)d_in[7], (const float*)d_in[9]  };
  const float* fp1_b[3] = { (const float*)d_in[6], (const float*)d_in[8], (const float*)d_in[10] };
  const float* fp2_w[3] = { (const float*)d_in[11], (const float*)d_in[13], (const float*)d_in[15] };
  const float* fp2_b[3] = { (const float*)d_in[12], (const float*)d_in[14], (const float*)d_in[16] };
  const float* gz_w = (const float*)d_in[17]; const float* gz_b = (const float*)d_in[18];
  const float* gr_w = (const float*)d_in[19]; const float* gr_b = (const float*)d_in[20];
  const float* gq_w = (const float*)d_in[21]; const float* gq_b = (const float*)d_in[22];

  // ---- carve workspace ----
  char* ws = (char*)d_ws;
  size_t off = 0;
  auto carve = [&](size_t bytes) -> void* {
    void* p = ws + off;
    off = (off + bytes + 255) & ~(size_t)255;
    return p;
  };
  const long long M1 = (long long)B_ * N_ * 16;   // 524288
  const long long M2 = (long long)B_ * N_ * 8;    // 262144
  const long long M3 = (long long)B_ * N_ * 4;    // 131072

  int*   idx   = (int*)  carve((size_t)B_ * N_ * 16 * sizeof(int));
  u16*   Abuf  = (u16*)  carve((size_t)(M3 * 224) * sizeof(u16));   // largest A (gates)
  float* Xbuf  = (float*)carve((size_t)(M1 * 32) * sizeof(float));  // largest X (fp1)
  u16*   Gnorm = (u16*)  carve((size_t)(M3 * 64) * sizeof(u16));    // gate norm output
  float* ff1   = (float*)carve((size_t)B_ * N_ * 32 * sizeof(float));
  float* ff2   = (float*)carve((size_t)B_ * N_ * 16 * sizeof(float));
  float* zpre  = (float*)carve((size_t)B_ * N_ * 64 * sizeof(float));
  float* rpre  = (float*)carve((size_t)B_ * N_ * 64 * sizeof(float));
  float* qpre  = (float*)carve((size_t)B_ * N_ * 64 * sizeof(float));
  float* rhb   = (float*)carve((size_t)B_ * N_ * 64 * sizeof(float));
  u32*   Wp    = (u32*)  carve((size_t)(224 / 32) * 4 * 256 * sizeof(u32));
  float* stats = (float*)carve((size_t)B_ * 64 * 2 * sizeof(float));
  (void)ws_size; (void)in_sizes; (void)n_in; (void)out_size;

  // ---- KNN ----
  knn16_kernel<<<dim3(N_ / 256, B_), 256, 0, stream>>>(pc, idx);

  // ---- fp1: 6 -> 32 -> 32 -> 32, k = 16 ----
  gather_fp1_kernel<<<(unsigned)((M1 + 255) / 256), 256, 0, stream>>>(pc, flow, idx, Abuf);
  layer_forward(Abuf, fp1_w[0], fp1_b[0], Wp, Xbuf, stats, Abuf, (int)M1,  6, 32, 32, 32, N_ * 16, 1, stream);
  layer_forward(Abuf, fp1_w[1], fp1_b[1], Wp, Xbuf, stats, Abuf, (int)M1, 32, 32, 32, 32, N_ * 16, 1, stream);
  layer_forward(Abuf, fp1_w[2], fp1_b[2], Wp, Xbuf, stats, Abuf, (int)M1, 32, 32, 32, 32, N_ * 16, 1, stream);
  {
    long long t = (long long)B_ * N_ * 32;
    maxk_kernel<<<(unsigned)((t + 255) / 256), 256, 0, stream>>>(Abuf, ff1, 16, 32, 32, t);
  }

  // ---- fp2: 35 -> 16 -> 16 -> 16, k = 8 ----
  gather_fp2_kernel<<<(unsigned)((M2 + 255) / 256), 256, 0, stream>>>(pc, ff1, idx, Abuf);
  layer_forward(Abuf, fp2_w[0], fp2_b[0], Wp, Xbuf, stats, Abuf, (int)M2, 35, 64, 16, 32, N_ * 8, 1, stream);
  layer_forward(Abuf, fp2_w[1], fp2_b[1], Wp, Xbuf, stats, Abuf, (int)M2, 16, 32, 16, 32, N_ * 8, 1, stream);
  layer_forward(Abuf, fp2_w[2], fp2_b[2], Wp, Xbuf, stats, Abuf, (int)M2, 16, 32, 16, 16, N_ * 8, 1, stream);
  {
    long long t = (long long)B_ * N_ * 16;
    maxk_kernel<<<(unsigned)((t + 255) / 256), 256, 0, stream>>>(Abuf, ff2, 8, 16, 16, t);
  }

  // ---- GRU gates: single 214 -> 64 layer each, k = 4 ----
  long long tg = (long long)B_ * N_ * 64;
  // z and r share the same gathered hx
  gather_gate_kernel<<<(unsigned)((M3 + 255) / 256), 256, 0, stream>>>(
      pc, h, 1, f1n, cost, ff2, flow, idx, Abuf);
  layer_forward(Abuf, gz_w, gz_b, Wp, Xbuf, stats, Gnorm, (int)M3, 214, 224, 64, 64, N_ * 4, 0, stream);
  maxk_kernel<<<(unsigned)((tg + 255) / 256), 256, 0, stream>>>(Gnorm, zpre, 4, 64, 64, tg);
  layer_forward(Abuf, gr_w, gr_b, Wp, Xbuf, stats, Gnorm, (int)M3, 214, 224, 64, 64, N_ * 4, 0, stream);
  maxk_kernel<<<(unsigned)((tg + 255) / 256), 256, 0, stream>>>(Gnorm, rpre, 4, 64, 64, tg);
  // rh = sigmoid(r) * h, then q gate
  rh_kernel<<<(unsigned)((tg + 255) / 256), 256, 0, stream>>>(rpre, h, rhb, tg);
  gather_gate_kernel<<<(unsigned)((M3 + 255) / 256), 256, 0, stream>>>(
      pc, rhb, 0, f1n, cost, ff2, flow, idx, Abuf);
  layer_forward(Abuf, gq_w, gq_b, Wp, Xbuf, stats, Gnorm, (int)M3, 214, 224, 64, 64, N_ * 4, 0, stream);
  maxk_kernel<<<(unsigned)((tg + 255) / 256), 256, 0, stream>>>(Gnorm, qpre, 4, 64, 64, tg);

  // ---- combine: out = (1-sigmoid(z))*h + sigmoid(z)*tanh(q) ----
  combine_kernel<<<(unsigned)((tg + 255) / 256), 256, 0, stream>>>(
      zpre, qpre, h, (float*)d_out, tg);
}
